// BitLinear_29540785062021
// MI455X (gfx1250) — compile-verified, matrix operand order checked by reading, not run on hardware
//
#include <hip/hip_runtime.h>
#include <hip/hip_bf16.h>
#include <stdint.h>

// ---------------------------------------------------------------------------
// BitLinear forward for MI455X (gfx1250, wave32, WMMA):
//   gamma = max(mean(|clip(W,-2,2)|), 1e-4)
//   Wq    = clip(round(clip(W)/gamma), -1, 1)   (ternary, exact in bf16)
//   out   = (Xbf16 @ Wq_bf16^T) * gamma + bias  (f32 accumulate via WMMA)
// GEMM: M=8192, N=4096, K=4096 -> compute-bound (AI ~ 820 FLOP/B); bf16 WMMA
// with double-buffered async global->LDS copies to hide HBM/L2 latency.
// ---------------------------------------------------------------------------

typedef __attribute__((ext_vector_type(16))) __bf16 bf16x16;
typedef __attribute__((ext_vector_type(8)))  __bf16 bf16x8;
typedef __attribute__((ext_vector_type(8)))  float  f32x8;

#ifndef __has_builtin
#define __has_builtin(x) 0
#endif

#if __has_builtin(__builtin_amdgcn_global_load_async_to_lds_b128) && \
    __has_builtin(__builtin_amdgcn_s_wait_asynccnt)
#define USE_ASYNC_COPY 1
// Builtin signature (from hipcc diagnostic): param0 = int4 AS(1)*, param1 = int4 AS(3)*
typedef __attribute__((__vector_size__(4 * sizeof(int)))) int v4i_b;
typedef __attribute__((address_space(1))) v4i_b gv4i;
typedef __attribute__((address_space(3))) v4i_b lv4i;
#else
#define USE_ASYNC_COPY 0
#endif

static constexpr int KDIM = 4096;   // D_IN
static constexpr int NDIM = 4096;   // D_OUT
static constexpr int BM = 128;
static constexpr int BN = 128;
static constexpr int BK = 64;
static constexpr int KSTEPS = KDIM / BK;   // 64
static constexpr int RED_BLOCKS = 1024;

// ---------------- Pass 1: per-block partial sums of |clip(w,-2,2)| ----------
__global__ void k_abs_partial(const float* __restrict__ w,
                              float* __restrict__ partial, int n4) {
    __shared__ float red[256];
    float s = 0.0f;
    const int stride = gridDim.x * blockDim.x;
    for (int i = blockIdx.x * blockDim.x + threadIdx.x; i < n4; i += stride) {
        float4 v = ((const float4*)w)[i];
        s += fminf(fabsf(v.x), 2.0f) + fminf(fabsf(v.y), 2.0f) +
             fminf(fabsf(v.z), 2.0f) + fminf(fabsf(v.w), 2.0f);
    }
    red[threadIdx.x] = s;
    __syncthreads();
    for (int off = 128; off > 0; off >>= 1) {
        if (threadIdx.x < off) red[threadIdx.x] += red[threadIdx.x + off];
        __syncthreads();
    }
    if (threadIdx.x == 0) partial[blockIdx.x] = red[0];
}

// ---------------- Pass 2: finalize gamma (deterministic, single block) ------
__global__ void k_gamma(const float* __restrict__ partial, int nPartial,
                        float* __restrict__ gammap, float invCount) {
    __shared__ float red[256];
    float s = 0.0f;
    for (int i = threadIdx.x; i < nPartial; i += 256) s += partial[i];
    red[threadIdx.x] = s;
    __syncthreads();
    for (int off = 128; off > 0; off >>= 1) {
        if (threadIdx.x < off) red[threadIdx.x] += red[threadIdx.x + off];
        __syncthreads();
    }
    if (threadIdx.x == 0) gammap[0] = fmaxf(red[0] * invCount, 1e-4f);
}

// ---------------- Pass 3: ternary-quantize W -> bf16 ------------------------
__global__ void k_quant_w(const float* __restrict__ w,
                          const float* __restrict__ gammap,
                          __hip_bfloat16* __restrict__ wq, int n4) {
    const float inv_g = 1.0f / gammap[0];
    const int i = blockIdx.x * blockDim.x + threadIdx.x;
    if (i >= n4) return;
    float4 v = ((const float4*)w)[i];
    float q[4] = {v.x, v.y, v.z, v.w};
    union { __hip_bfloat16 h[4]; uint2 u; } pk;
#pragma unroll
    for (int c = 0; c < 4; ++c) {
        float wc = fminf(fmaxf(q[c], -2.0f), 2.0f);
        float r  = rintf(wc * inv_g);               // RNE, matches jnp.round
        r = fminf(fmaxf(r, -1.0f), 1.0f);           // ternary {-1,0,1}: exact bf16
        pk.h[c] = __float2bfloat16(r);
    }
    ((uint2*)wq)[i] = pk.u;
}

// ---------------- Pass 4: X f32 -> bf16 -------------------------------------
__global__ void k_cvt_x(const float* __restrict__ x,
                        __hip_bfloat16* __restrict__ xb, int n4) {
    const int i = blockIdx.x * blockDim.x + threadIdx.x;
    if (i >= n4) return;
    float4 v = ((const float4*)x)[i];
    union { __hip_bfloat16 h[4]; uint2 u; } pk;
    pk.h[0] = __float2bfloat16(v.x);
    pk.h[1] = __float2bfloat16(v.y);
    pk.h[2] = __float2bfloat16(v.z);
    pk.h[3] = __float2bfloat16(v.w);
    ((uint2*)xb)[i] = pk.u;
}

// ---------------- Pass 5: bf16 WMMA GEMM (double-buffered async) ------------
// out[m][n] = gamma * sum_k Xb[m][k] * Wq[n][k] + bias[n]
// Block: 256 threads = 8 waves; tile 128x128x64, double-buffered LDS (64 KB).
// Waves: wm in 0..3 (32 rows each), wn in 0..1 (64 cols each)
//   -> each wave: 2 (M) x 4 (N) tiles of 16x16; 16 v_wmma per K-iteration.
__global__ __launch_bounds__(256)
void k_gemm(const __hip_bfloat16* __restrict__ A,   // [M][K] bf16
            const __hip_bfloat16* __restrict__ W,   // [N][K] bf16 ternary
            const float* __restrict__ bias,
            const float* __restrict__ gammap,
            float* __restrict__ out) {
    __shared__ __align__(16) __hip_bfloat16 As[2][BM][BK];  // 2 x 16 KB
    __shared__ __align__(16) __hip_bfloat16 Ws[2][BN][BK];  // 2 x 16 KB

    const int tid  = threadIdx.x;
    const int lane = tid & 31;
    const int wave = tid >> 5;
    const int wm   = wave & 3;    // 0..3
    const int wn   = wave >> 2;   // 0..1
    const int m0   = blockIdx.y * BM;
    const int n0   = blockIdx.x * BN;

    // Loader mapping: 128 rows x 64 cols bf16 per tile; each of 256 threads
    // moves 32 bf16 (four b128 chunks): row = tid/2, col base = (tid&1)*32.
    const int lrow = tid >> 1;
    const int lcol = (tid & 1) * 32;

    // WMMA 16-bit A/B fragment lane pattern (ISA 7.12.2):
    //   lanes 0-15 : row/col = lane,    K chunks {0..7, 16..23}
    //   lanes 16-31: row/col = lane-16, K chunks {8..15, 24..31}
    const int laneRC = lane & 15;
    const int laneK  = (lane >> 4) * 8;   // 0 or 8

    const float gamma = gammap[0];

    f32x8 acc[2][4] = {};

    // Issue one tile-pair's async copies (8 x b128 per thread).
    auto issue_tile = [&](int k0, int buf) {
        const size_t aIdx = (size_t)(m0 + lrow) * KDIM + k0 + lcol;
        const size_t wIdx = (size_t)(n0 + lrow) * KDIM + k0 + lcol;
#if USE_ASYNC_COPY
#pragma unroll
        for (int c = 0; c < 4; ++c)
            __builtin_amdgcn_global_load_async_to_lds_b128(
                (gv4i*)(A + aIdx + 8 * c),
                (lv4i*)&As[buf][lrow][lcol + 8 * c], 0, 0);
#pragma unroll
        for (int c = 0; c < 4; ++c)
            __builtin_amdgcn_global_load_async_to_lds_b128(
                (gv4i*)(W + wIdx + 8 * c),
                (lv4i*)&Ws[buf][lrow][lcol + 8 * c], 0, 0);
#else
#pragma unroll
        for (int c = 0; c < 4; ++c) {
            *(bf16x8*)&As[buf][lrow][lcol + 8 * c] = *(const bf16x8*)(A + aIdx + 8 * c);
            *(bf16x8*)&Ws[buf][lrow][lcol + 8 * c] = *(const bf16x8*)(W + wIdx + 8 * c);
        }
#endif
    };

    issue_tile(0, 0);   // prologue: fill buffer 0

    for (int k = 0; k < KSTEPS; ++k) {
        const int buf = k & 1;
        if (k + 1 < KSTEPS) {
            issue_tile((k + 1) * BK, buf ^ 1);   // prefetch next tile into alt buffer
#if USE_ASYNC_COPY
            // 8 newest outstanding = next tile; older 8 (current tile, issued a
            // full iteration ago) are complete since async loads retire in order.
            __builtin_amdgcn_s_wait_asynccnt(8);
#endif
        } else {
#if USE_ASYNC_COPY
            __builtin_amdgcn_s_wait_asynccnt(0);
#endif
        }
        __syncthreads();   // current buffer ready across all waves

        if (k + 2 < KSTEPS) {
            __builtin_prefetch(A + (size_t)(m0 + lrow) * KDIM + (k + 2) * BK + lcol, 0, 3);
            __builtin_prefetch(W + (size_t)(n0 + lrow) * KDIM + (k + 2) * BK + lcol, 0, 3);
        }

#pragma unroll
        for (int kk = 0; kk < BK; kk += 32) {
            bf16x16 afrag[2];
#pragma unroll
            for (int i = 0; i < 2; ++i) {
                const int r = wm * 32 + i * 16 + laneRC;
                ((bf16x8*)&afrag[i])[0] = *(const bf16x8*)&As[buf][r][kk + laneK];
                ((bf16x8*)&afrag[i])[1] = *(const bf16x8*)&As[buf][r][kk + laneK + 16];
            }
            bf16x16 bfrag[4];
#pragma unroll
            for (int j = 0; j < 4; ++j) {
                const int c = wn * 64 + j * 16 + laneRC;
                ((bf16x8*)&bfrag[j])[0] = *(const bf16x8*)&Ws[buf][c][kk + laneK];
                ((bf16x8*)&bfrag[j])[1] = *(const bf16x8*)&Ws[buf][c][kk + laneK + 16];
            }

#pragma unroll
            for (int i = 0; i < 2; ++i)
#pragma unroll
                for (int j = 0; j < 4; ++j)
                    acc[i][j] = __builtin_amdgcn_wmma_f32_16x16x32_bf16(
                        false, afrag[i], false, bfrag[j],
                        (short)0, acc[i][j], false, false);
        }

        __syncthreads();   // protect alt buffer: no refill while still being read
    }

    // Epilogue. 16x16 f32 C/D layout: VGPR r -> (M=r, N=lane) for lanes 0-15,
    // (M=8+r, N=lane-16) for lanes 16-31.
    const int cmBase = wm * 32 + (lane >> 4) * 8;
    const int cnBase = wn * 64 + (lane & 15);
#pragma unroll
    for (int i = 0; i < 2; ++i) {
#pragma unroll
        for (int j = 0; j < 4; ++j) {
            const int n = n0 + cnBase + j * 16;
            const float b = bias[n];
#pragma unroll
            for (int r = 0; r < 8; ++r) {
                const int m = m0 + cmBase + i * 16 + r;
                out[(size_t)m * NDIM + n] = acc[i][j][r] * gamma + b;
            }
        }
    }
}

// ---------------------------------------------------------------------------
extern "C" void kernel_launch(void* const* d_in, const int* in_sizes, int n_in,
                              void* d_out, int out_size, void* d_ws, size_t ws_size,
                              hipStream_t stream) {
    const float* x    = (const float*)d_in[0];   // [B,S,K] f32
    const float* w    = (const float*)d_in[1];   // [N,K]   f32
    const float* bias = (const float*)d_in[2];   // [N]     f32
    float*       out  = (float*)d_out;           // [B,S,N] f32

    const int nX = in_sizes[0];                  // M*K = 33,554,432
    const int nW = in_sizes[1];                  // N*K = 16,777,216
    const int M  = nX / KDIM;                    // 8192

    // Workspace layout:
    //   [0,4)            : gamma (f32)
    //   [256, 256+4K)    : reduction partials
    //   [64K, 64K+32M)   : Wq bf16  (N*K*2 bytes)
    //   [.., ..+64M)     : Xb bf16  (M*K*2 bytes)
    char* ws = (char*)d_ws;
    float*          gammap  = (float*)ws;
    float*          partial = (float*)(ws + 256);
    __hip_bfloat16* wq      = (__hip_bfloat16*)(ws + 65536);
    __hip_bfloat16* xb      = (__hip_bfloat16*)(ws + 65536 + (size_t)NDIM * KDIM * 2);

    k_abs_partial<<<RED_BLOCKS, 256, 0, stream>>>(w, partial, nW / 4);
    k_gamma<<<1, 256, 0, stream>>>(partial, RED_BLOCKS, gammap, 1.0f / (float)nW);

    const int nW4 = nW / 4;
    k_quant_w<<<(nW4 + 255) / 256, 256, 0, stream>>>(w, gammap, wq, nW4);
    const int nX4 = nX / 4;
    k_cvt_x<<<(nX4 + 255) / 256, 256, 0, stream>>>(x, xb, nX4);

    dim3 grid(NDIM / BN, M / BM);                // (32, 64)
    k_gemm<<<grid, 256, 0, stream>>>(xb, wq, bias, gammap, out);
}
